// Qwen3MoeRouter_22892175688404
// MI455X (gfx1250) — matrix operand order repaired and checked by reading, not compiled, not run on hardware
//
#include <hip/hip_runtime.h>
#include <hip/hip_bf16.h>
#include <stdint.h>

typedef __attribute__((ext_vector_type(16))) __bf16 v16bf;
typedef __attribute__((ext_vector_type(8)))  __bf16 v8bf;
typedef __attribute__((ext_vector_type(4)))  __bf16 v4bf;
typedef __attribute__((ext_vector_type(8)))  float  v8f;
typedef __attribute__((ext_vector_type(4)))  float  f4;

constexpr int NT   = 16384;   // tokens
constexpr int DIM  = 4096;    // hidden
constexpr int NE   = 128;     // experts
constexpr int TOPK = 8;
constexpr int KCHUNK = 64;    // K staging chunk (2 WMMA k-steps)
constexpr int NPAIR = NT * TOPK;        // 131072
constexpr int NTILE = NPAIR / 256;      // 512 sort tiles

// ---- CDNA5 async global->LDS path (guarded; falls back to sync copy) ----
#if defined(__gfx1250__) && __has_builtin(__builtin_amdgcn_global_load_async_to_lds_b128)
#define HAVE_ASYNC 1
#else
#define HAVE_ASYNC 0
#endif

#if defined(__gfx1250__) && __has_builtin(__builtin_amdgcn_s_wait_asynccnt)
#define WAIT_ASYNC(n) __builtin_amdgcn_s_wait_asynccnt(n)
#else
#define WAIT_ASYNC(n) asm volatile("s_wait_asynccnt %0" :: "i"(n) : "memory")
#endif

// builtin's parameter type: int __attribute__((vector_size(16)))
typedef int b128v __attribute__((vector_size(16)));
typedef __attribute__((address_space(1))) b128v* as1_b128p;
typedef __attribute__((address_space(3))) b128v* as3_b128p;

__device__ __forceinline__ void stage16B(float* lds_dst, const float* gsrc) {
#if HAVE_ASYNC
  // generic->AS3: LDS generic addr low 32 bits are the LDS offset
  __builtin_amdgcn_global_load_async_to_lds_b128(
      (as1_b128p)(uintptr_t)gsrc,
      (as3_b128p)(uint32_t)(uintptr_t)lds_dst,
      0, 0);
#else
  *(f4*)lds_dst = *(const f4*)gsrc;
#endif
}

__device__ __forceinline__ v8f wmma_bf16(v16bf a, v16bf b, v8f c) {
  return __builtin_amdgcn_wmma_f32_16x16x32_bf16(false, a, false, b, (short)0, c,
                                                 false, false);
}

// =====================================================================
// K0: one-time gate_w f32 -> bf16 hi/lo split (L2-resident, 2MB read)
// =====================================================================
__global__ __launch_bounds__(256) void conv_gw(
    const float* __restrict__ gw, __bf16* __restrict__ gwh,
    __bf16* __restrict__ gwl) {
  const int i = (blockIdx.x * 256 + threadIdx.x) * 4;
  f4 v = *(const f4*)(gw + i);
  v4bf h, l;
#pragma unroll
  for (int j = 0; j < 4; ++j) {
    __bf16 hh = (__bf16)v[j];
    h[j] = hh;
    l[j] = (__bf16)(v[j] - (float)hh);
  }
  *(v4bf*)(gwh + i) = h;
  *(v4bf*)(gwl + i) = l;
}

// =====================================================================
// K1: scores[t][e] = sum_k x[t][k] * gw[e][k]   (bf16x3 split via WMMA)
// block: 256 thr = 8 waves; covers 16 tokens x all 128 experts so x is
// read from HBM exactly once.  x chunk staged via async global->LDS,
// converted to bf16 hi/lo in LDS once per element per block.
// =====================================================================
__global__ __launch_bounds__(256) void router_gemm(
    const float* __restrict__ x, const __bf16* __restrict__ gwh,
    const __bf16* __restrict__ gwl, float* __restrict__ scores) {
  __shared__ float  sF[2][16 * KCHUNK];   // raw staged f32 (2 x 4KB)
  __shared__ __bf16 sH[2][16 * KCHUNK];   // A hi (2 x 2KB)
  __shared__ __bf16 sL[2][16 * KCHUNK];   // A lo (2 x 2KB)

  const int tid  = threadIdx.x;
  const int lane = tid & 31;
  const int wave = tid >> 5;
  const int col  = lane & 15;            // A row (M) and B column (expert)
  const bool hiL = lane >= 16;
  const int t0   = blockIdx.x * 16;
  const int e0   = wave * 16;

  // staging: each thread owns 4 consecutive floats (16B) of the chunk
  const int srow = tid >> 4;             // 0..15
  const int skof = (tid & 15) * 4;       // 0..60
  const float* gsrc = x + (size_t)(t0 + srow) * DIM + skof;
  const int own = tid * 4;               // flat offset of owned 16B region

  v8f acc = {};

  stage16B(&sF[0][own], gsrc + 0);       // prologue: chunk 0 in flight

  for (int kc = 0, it = 0; kc < DIM; kc += KCHUNK, ++it) {
    const int buf = it & 1;
    const bool more = (kc + KCHUNK) < DIM;

    WAIT_ASYNC(0);                       // my own 16B of chunk `it` landed
    {                                    // convert ONLY my own region
      f4 v = *(const f4*)&sF[buf][own];
      v4bf h, l;
#pragma unroll
      for (int j = 0; j < 4; ++j) {
        __bf16 hh = (__bf16)v[j];
        h[j] = hh;
        l[j] = (__bf16)(v[j] - (float)hh);
      }
      *(v4bf*)&sH[buf][own] = h;
      *(v4bf*)&sL[buf][own] = l;
    }
    if (more)                            // prefetch next chunk (other buffer)
      stage16B(&sF[buf ^ 1][own], gsrc + kc + KCHUNK);
    __syncthreads();                     // sH/sL[buf] visible to all waves

#pragma unroll
    for (int sub = 0; sub < 2; ++sub) {
      // A tile: lanes 0-15 -> K {0..15}, lanes 16-31 -> K {16..31}
      const int ab = col * KCHUNK + sub * 32 + (hiL ? 16 : 0);
      v8bf ah0 = *(const v8bf*)&sH[buf][ab];
      v8bf ah1 = *(const v8bf*)&sH[buf][ab + 8];
      v8bf al0 = *(const v8bf*)&sL[buf][ab];
      v8bf al1 = *(const v8bf*)&sL[buf][ab + 8];
      v16bf a_h = __builtin_shufflevector(ah0, ah1, 0,1,2,3,4,5,6,7,8,9,10,11,12,13,14,15);
      v16bf a_l = __builtin_shufflevector(al0, al1, 0,1,2,3,4,5,6,7,8,9,10,11,12,13,14,15);
      // B tile: lane holds column e0+col, 16 consecutive K (contiguous bf16)
      const size_t bo = (size_t)(e0 + col) * DIM + kc + sub * 32 + (hiL ? 16 : 0);
      v16bf b_h = *(const v16bf*)(gwh + bo);
      v16bf b_l = *(const v16bf*)(gwl + bo);
      // bf16x3 split: (Ah+Al)(Bh+Bl) ~= AhBh + AhBl + AlBh
      acc = wmma_bf16(a_h, b_h, acc);
      acc = wmma_bf16(a_h, b_l, acc);
      acc = wmma_bf16(a_l, b_h, acc);
    }
    __syncthreads();                     // protect sH/sL[buf] until all read
  }

  // D layout: VGPR r -> M = r + (lane>=16 ? 8 : 0), N = lane&15
#pragma unroll
  for (int r = 0; r < 8; ++r) {
    const int token = t0 + r + (hiL ? 8 : 0);
    scores[(size_t)token * NE + e0 + col] = acc[r];
  }
}

// =====================================================================
// K2: softmax over 128 experts + iterative top-8 (stable ties) + renorm
// one wave per token; lane l owns experts {l, l+32, l+64, l+96}
// =====================================================================
__global__ __launch_bounds__(256) void router_topk(
    const float* __restrict__ scores, float* __restrict__ weights,
    int* __restrict__ eids) {
  const int lane = threadIdx.x & 31;
  const int wave = threadIdx.x >> 5;
  const int t = blockIdx.x * 8 + wave;
  const float* s = scores + (size_t)t * NE;

  float s0 = s[lane], s1 = s[lane + 32], s2 = s[lane + 64], s3 = s[lane + 96];

  float m = fmaxf(fmaxf(s0, s1), fmaxf(s2, s3));
#pragma unroll
  for (int off = 16; off >= 1; off >>= 1) m = fmaxf(m, __shfl_xor(m, off));

  float e0 = __expf(s0 - m), e1 = __expf(s1 - m), e2 = __expf(s2 - m), e3 = __expf(s3 - m);
  float sum = e0 + e1 + e2 + e3;
#pragma unroll
  for (int off = 16; off >= 1; off >>= 1) sum += __shfl_xor(sum, off);
  const float inv = 1.0f / sum;
  float p0 = e0 * inv, p1 = e1 * inv, p2 = e2 * inv, p3 = e3 * inv;

  float myw = 0.f; int mye = 0; float acc8 = 0.f;
#pragma unroll
  for (int it = 0; it < TOPK; ++it) {
    float bv = p0; int bi = lane;
    if (p1 > bv) { bv = p1; bi = lane + 32; }
    if (p2 > bv) { bv = p2; bi = lane + 64; }
    if (p3 > bv) { bv = p3; bi = lane + 96; }
#pragma unroll
    for (int off = 16; off >= 1; off >>= 1) {
      float ov = __shfl_xor(bv, off);
      int   oi = __shfl_xor(bi, off);
      if (ov > bv || (ov == bv && oi < bi)) { bv = ov; bi = oi; }
    }
    acc8 += bv;
    if (lane == it) { myw = bv; mye = bi; }
    if ((bi & 31) == lane) {           // knock out the winner
      const int slot = bi >> 5;
      if      (slot == 0) p0 = -1.f;
      else if (slot == 1) p1 = -1.f;
      else if (slot == 2) p2 = -1.f;
      else                p3 = -1.f;
    }
  }
  const float denom = fmaxf(acc8, 1e-8f);
  if (lane < TOPK) {
    weights[(size_t)t * TOPK + lane] = myw / denom;
    eids[(size_t)t * TOPK + lane] = mye;
  }
}

// =====================================================================
// K3: per-tile expert histograms (tile = 256 consecutive pairs)
// =====================================================================
__global__ __launch_bounds__(256) void tile_hist(
    const int* __restrict__ eids, int* __restrict__ hist) {
  __shared__ int h[NE];
  const int tid = threadIdx.x;
  if (tid < NE) h[tid] = 0;
  __syncthreads();
  const int e = eids[blockIdx.x * 256 + tid];
  atomicAdd(&h[e], 1);
  __syncthreads();
  if (tid < NE) hist[blockIdx.x * NE + tid] = h[tid];
}

// =====================================================================
// K4: per-expert exclusive scan over tiles (in place) + expert base scan
//     + write counts (int32) to output
// =====================================================================
__global__ __launch_bounds__(128) void scan_kernel(
    int* __restrict__ hist, int* __restrict__ ebase, int* __restrict__ out_cnt) {
  __shared__ int sc[NE];
  const int e = threadIdx.x;
  int run = 0;
  for (int t = 0; t < NTILE; ++t) {
    const int v = hist[t * NE + e];
    hist[t * NE + e] = run;
    run += v;
  }
  out_cnt[e] = run;
  sc[e] = run;
  __syncthreads();
  int total = run;
#pragma unroll
  for (int off = 1; off < NE; off <<= 1) {
    int y = (e >= off) ? sc[e - off] : 0;
    __syncthreads();
    sc[e] = (total += y);
    __syncthreads();
  }
  ebase[e] = sc[e] - run;   // exclusive prefix of counts
}

// =====================================================================
// K5: stable scatter: pos = ebase[e] + tile_prefix[tile][e] + local rank
// =====================================================================
__global__ __launch_bounds__(256) void scatter_kernel(
    const int* __restrict__ eids, const float* __restrict__ weights,
    const int* __restrict__ hist, const int* __restrict__ ebase,
    float* __restrict__ out_w, int* __restrict__ out_tok) {
  __shared__ int se[256];
  const int tid = threadIdx.x;
  const int i = blockIdx.x * 256 + tid;
  const int e = eids[i];
  const float w = weights[i];
  se[tid] = e;
  __syncthreads();
  int rank = 0;
  for (int j = 0; j < tid; ++j) rank += (se[j] == e);
  const int pos = ebase[e] + hist[blockIdx.x * NE + e] + rank;
  out_w[pos] = w;
  out_tok[pos] = i >> 3;   // token index = flat / TOPK
}

// =====================================================================
extern "C" void kernel_launch(void* const* d_in, const int* in_sizes, int n_in,
                              void* d_out, int out_size, void* d_ws, size_t ws_size,
                              hipStream_t stream) {
  const float* x  = (const float*)d_in[0];
  const float* gw = (const float*)d_in[1];

  float*  scores  = (float*)d_ws;                      // NT*NE f32
  float*  weights = scores + (size_t)NT * NE;          // NPAIR f32
  int*    eids    = (int*)(weights + NPAIR);           // NPAIR i32
  int*    hist    = eids + NPAIR;                      // NTILE*NE i32
  int*    ebase   = hist + NTILE * NE;                 // NE i32
  __bf16* gwh     = (__bf16*)(ebase + NE);             // NE*DIM bf16 (hi)
  __bf16* gwl     = gwh + (size_t)NE * DIM;            // NE*DIM bf16 (lo)

  float* out_w   = (float*)d_out;                      // NPAIR f32
  int*   out_tok = (int*)d_out + NPAIR;                // NPAIR i32
  int*   out_cnt = (int*)d_out + 2 * NPAIR;            // NE i32

  conv_gw       <<<NE * DIM / 1024, 256, 0, stream>>>(gw, gwh, gwl);
  router_gemm   <<<NT / 16,         256, 0, stream>>>(x, gwh, gwl, scores);
  router_topk   <<<NT / 8,          256, 0, stream>>>(scores, weights, eids);
  tile_hist     <<<NTILE,           256, 0, stream>>>(eids, hist);
  scan_kernel   <<<1,               128, 0, stream>>>(hist, ebase, out_cnt);
  scatter_kernel<<<NTILE,           256, 0, stream>>>(eids, weights, hist, ebase, out_w, out_tok);
}
